// GroupedQueryAttention_1142461301302
// MI455X (gfx1250) — matrix operand order repaired and checked by reading, not compiled, not run on hardware
//
#include <hip/hip_runtime.h>

// ---------------- problem constants ----------------
#define S_LEN    2048
#define D_MODEL  4096
#define N_HEADS  32
#define N_KV     16
#define HEAD_DIM 128
#define D_KV     (N_KV * HEAD_DIM)   // 2048

typedef __attribute__((ext_vector_type(16))) __bf16 v16bf;
typedef __attribute__((ext_vector_type(8)))  float  v8f;
// gcc-style vectors to match builtin parameter declarations
typedef int          vec4i  __attribute__((vector_size(16)));
typedef unsigned int tdm_u4 __attribute__((vector_size(16)));
typedef int          tdm_i8 __attribute__((vector_size(32)));
typedef int          tdm_i4 __attribute__((vector_size(16)));

// CDNA5 data-mover feature detection (compile-safe on any toolchain)
#if defined(__gfx1250__) && __has_builtin(__builtin_amdgcn_global_load_async_to_lds_b128) && __has_builtin(__builtin_amdgcn_s_wait_asynccnt)
#define HAVE_ASYNC 1
#else
#define HAVE_ASYNC 0
#endif
#if defined(__gfx1250__) && __has_builtin(__builtin_amdgcn_tensor_load_to_lds) && __has_builtin(__builtin_amdgcn_s_wait_tensorcnt)
#define HAVE_TDM 1
#else
#define HAVE_TDM 0
#endif

#if HAVE_ASYNC
#define AS1_V4I(p) ((__attribute__((address_space(1))) vec4i*)(p))
#define AS3_V4I(p) ((__attribute__((address_space(3))) vec4i*)(p))
#endif

struct Frag { uint4 q[2]; };   // 32 bytes = 16 bf16 = one WMMA A/B operand (wave32)

__device__ __forceinline__ unsigned short f2bf(float f) {
  unsigned u = __builtin_bit_cast(unsigned, f);
  u += 0x7FFFu + ((u >> 16) & 1u);          // round-to-nearest-even
  return (unsigned short)(u >> 16);
}

__device__ __forceinline__ v8f wmma_bf16(const Frag& a, const Frag& b, v8f c) {
  return __builtin_amdgcn_wmma_f32_16x16x32_bf16(
      false, __builtin_bit_cast(v16bf, a),
      false, __builtin_bit_cast(v16bf, b),
      (short)0, c, false, false);
}

__device__ __forceinline__ float rmax16(float x) {
  x = fmaxf(x, __shfl_xor(x, 1, 32));
  x = fmaxf(x, __shfl_xor(x, 2, 32));
  x = fmaxf(x, __shfl_xor(x, 4, 32));
  x = fmaxf(x, __shfl_xor(x, 8, 32));
  return x;
}
__device__ __forceinline__ float rsum16(float x) {
  x += __shfl_xor(x, 1, 32);
  x += __shfl_xor(x, 2, 32);
  x += __shfl_xor(x, 4, 32);
  x += __shfl_xor(x, 8, 32);
  return x;
}

#if HAVE_TDM
__device__ __forceinline__ unsigned lds_offset_of(const void* p) {
  return (unsigned)(unsigned long long)(__attribute__((address_space(3))) const void*)p;
}
#endif

// ---------------- f32 -> bf16 convert ----------------
__global__ void cvt_kernel(const float* __restrict__ in,
                           unsigned short* __restrict__ out, int n) {
  int i = blockIdx.x * 256 + threadIdx.x;
  if (i < n) out[i] = f2bf(in[i]);
}

// ---------------- RoPE (f32 in, bf16 out) ----------------
__global__ void rope_kernel(const float* __restrict__ Xf,
                            const float* __restrict__ cosT,
                            const float* __restrict__ sinT,
                            unsigned short* __restrict__ out, int heads) {
  int idx = blockIdx.x * 256 + threadIdx.x;
  int total = S_LEN * heads * 64;
  if (idx >= total) return;
  int d  = idx & 63;
  int t  = idx >> 6;
  int hh = t % heads;
  int s  = t / heads;
  int base = s * heads * HEAD_DIM + hh * HEAD_DIM;
  float q1 = Xf[base + d], q2 = Xf[base + d + 64];
  float c1 = cosT[s * HEAD_DIM + d],      s1 = sinT[s * HEAD_DIM + d];
  float c2 = cosT[s * HEAD_DIM + d + 64], s2 = sinT[s * HEAD_DIM + d + 64];
  out[base + d]      = f2bf(q1 * c1 - q2 * s1);
  out[base + d + 64] = f2bf(q2 * c2 + q1 * s2);
}

// ---------------- bf16 GEMM: C[M,N] = A[M,K] * B[N,K]^T ----------------
// block = 128x128 tile, 256 threads = 8 waves, wave tile = 32x64 (2x4 WMMA tiles)
// Double-buffered LDS (one barrier per K-step); staging via async-to-LDS when available.
__global__ __launch_bounds__(256) void gemm_bf16_kernel(
    const unsigned short* __restrict__ A, const unsigned short* __restrict__ B,
    void* __restrict__ C, int K, int N, int store_bf16) {
  __shared__ __align__(16) unsigned short As[2][128 * 40]; // stride 40 halves = 20 banks/row
  __shared__ __align__(16) unsigned short Bs[2][128 * 40];
  const int bm = blockIdx.y * 128;
  const int bn = blockIdx.x * 128;
  const int w  = threadIdx.x >> 5, lane = threadIdx.x & 31;
  const int wy = w >> 1, wx = w & 1;
  const int r  = lane & 15, hi = lane >> 4;
  v8f acc[2][4] = {};

#if HAVE_ASYNC
#define GEMM_STAGE(buf, koff)                                                        \
  for (int i_ = 0; i_ < 2; ++i_) {                                                   \
    int c_ = threadIdx.x + i_ * 256;                                                 \
    int row_ = c_ >> 2, seg_ = c_ & 3;                                               \
    __builtin_amdgcn_global_load_async_to_lds_b128(                                  \
        AS1_V4I(A + (size_t)(bm + row_) * K + (koff) + seg_ * 8),                    \
        AS3_V4I(&As[buf][row_ * 40 + seg_ * 8]), 0, 0);                              \
    __builtin_amdgcn_global_load_async_to_lds_b128(                                  \
        AS1_V4I(B + (size_t)(bn + row_) * K + (koff) + seg_ * 8),                    \
        AS3_V4I(&Bs[buf][row_ * 40 + seg_ * 8]), 0, 0);                              \
  }
#else
#define GEMM_STAGE(buf, koff)                                                        \
  for (int i_ = 0; i_ < 2; ++i_) {                                                   \
    int c_ = threadIdx.x + i_ * 256;                                                 \
    int row_ = c_ >> 2, seg_ = c_ & 3;                                               \
    *(uint4*)&As[buf][row_ * 40 + seg_ * 8] =                                        \
        *(const uint4*)&A[(size_t)(bm + row_) * K + (koff) + seg_ * 8];              \
    *(uint4*)&Bs[buf][row_ * 40 + seg_ * 8] =                                        \
        *(const uint4*)&B[(size_t)(bn + row_) * K + (koff) + seg_ * 8];              \
  }
#endif

  GEMM_STAGE(0, 0)
  for (int k0 = 0; k0 < K; k0 += 32) {
    const int buf = (k0 >> 5) & 1;
#if HAVE_ASYNC
    __builtin_amdgcn_s_wait_asynccnt(0);
#endif
    __syncthreads();
    if (k0 + 32 < K) { GEMM_STAGE(buf ^ 1, k0 + 32) }

    Frag a[2], b[4];
#pragma unroll
    for (int mt = 0; mt < 2; ++mt) {      // A layout: lane=row, K split across halves
      const unsigned short* p = &As[buf][(wy * 32 + mt * 16 + r) * 40];
      a[mt].q[0] = *(const uint4*)(p + hi * 8);
      a[mt].q[1] = *(const uint4*)(p + 16 + hi * 8);
    }
#pragma unroll
    for (int nt = 0; nt < 4; ++nt) {      // B layout: lane=col, K split across halves
      const unsigned short* p = &Bs[buf][(wx * 64 + nt * 16 + r) * 40];
      b[nt].q[0] = *(const uint4*)(p + hi * 8);
      b[nt].q[1] = *(const uint4*)(p + 16 + hi * 8);
    }
#pragma unroll
    for (int mt = 0; mt < 2; ++mt)
#pragma unroll
      for (int nt = 0; nt < 4; ++nt)
        acc[mt][nt] = wmma_bf16(a[mt], b[nt], acc[mt][nt]);
  }
#pragma unroll
  for (int mt = 0; mt < 2; ++mt)
#pragma unroll
    for (int nt = 0; nt < 4; ++nt) {
      int col = bn + wx * 64 + nt * 16 + r;
#pragma unroll
      for (int j = 0; j < 8; ++j) {
        int rowm = bm + wy * 32 + mt * 16 + hi * 8 + j;  // C layout: vgpr j = row
        float v = acc[mt][nt][j];
        if (store_bf16)
          ((unsigned short*)C)[(size_t)rowm * N + col] = f2bf(v);
        else
          ((float*)C)[(size_t)rowm * N + col] = v;
      }
    }
}

// ---------------- flash attention (causal, GQA groups=2) ----------------
// grid = (S/128, H); block = 256 (8 waves); wave owns 16 query rows
__global__ __launch_bounds__(256) void attn_kernel(
    const unsigned short* __restrict__ Q, const unsigned short* __restrict__ Kv,
    const unsigned short* __restrict__ Vv, unsigned short* __restrict__ O) {
  __shared__ __align__(16) unsigned short Kt[64 * 136];   // [key][hd], 272B padded rows
  __shared__ __align__(16) unsigned short Vt[128 * 72];   // [hd][key], transposed
  __shared__ __align__(16) unsigned short Ps[8 * 16 * 72];// per-wave P scratch
  const int h     = blockIdx.y;
  const int kvh   = h >> 1;                // GROUPS = 2
  const int qbase = blockIdx.x * 128;
  const int w = threadIdx.x >> 5, lane = threadIdx.x & 31;
  const int r = lane & 15, hi = lane >> 4;
  const int qr = qbase + w * 16;

  // Q fragments (16 rows x 128 hd) kept in registers
  Frag qa[4];
  {
    const unsigned short* qp = Q + (size_t)(qr + r) * D_MODEL + h * HEAD_DIM;
#pragma unroll
    for (int kc = 0; kc < 4; ++kc) {
      qa[kc].q[0] = *(const uint4*)(qp + kc * 32 + hi * 8);
      qa[kc].q[1] = *(const uint4*)(qp + kc * 32 + 16 + hi * 8);
    }
  }
  v8f o[8] = {};
  float m[8], l[8];
#pragma unroll
  for (int j = 0; j < 8; ++j) { m[j] = -1e30f; l[j] = 0.f; }

  const int nkb = blockIdx.x * 2 + 2;      // causal: key blocks 0 .. (qmax/64)
  unsigned short* pw = &Ps[w * 16 * 72];

  for (int kb = 0; kb < nkb; ++kb) {
    __syncthreads();
    {   // ---- stage K tile into LDS ----
#if HAVE_TDM
      if (w == 0) {
        // Tensor Data Mover: 2-byte elements, 128 x 64 tile, row stride 2048 elems,
        // LDS padding: insert 4 DWORDs after every 64 DWORDs (256B row -> 272B row).
        unsigned long long ga =
            (unsigned long long)(Kv + (size_t)(kb * 64) * D_KV + kvh * HEAD_DIM);
        tdm_u4 g0 = { 1u,                              // count=1 (valid descriptor)
                      lds_offset_of(Kt),               // lds_addr
                      (unsigned)ga,                    // global_addr[31:0]
                      (unsigned)(ga >> 32) | 0x80000000u }; // addr[56:32] | type=2
        tdm_i8 g1 = { (1 << 16) | (1 << 20) | (5 << 22) | (3 << 25), // 2B, pad 64DW+4DW
                      128 << 16,                       // tensor_dim0 = 128
                      64 << 16,                        // tensor_dim1 = 64
                      128 << 16,                       // tile_dim0  = 128
                      64,                              // tile_dim1  = 64
                      D_KV,                            // tensor_dim0_stride = 2048
                      0, 0 };
        tdm_i4 gz4 = { 0, 0, 0, 0 };
        tdm_i8 gz8 = { 0, 0, 0, 0, 0, 0, 0, 0 };
        __builtin_amdgcn_tensor_load_to_lds(g0, g1, gz4, gz4, gz8, 0);
        __builtin_amdgcn_s_wait_tensorcnt(0);
      }
#elif HAVE_ASYNC
      {
        int row = threadIdx.x >> 2, seg = threadIdx.x & 3;
        const unsigned short* ks =
            Kv + (size_t)(kb * 64 + row) * D_KV + kvh * HEAD_DIM + seg * 32;
#pragma unroll
        for (int i = 0; i < 4; ++i)
          __builtin_amdgcn_global_load_async_to_lds_b128(
              AS1_V4I(ks + i * 8),
              AS3_V4I(&Kt[row * 136 + seg * 32 + i * 8]), 0, 0);
      }
#else
      {
        int row = threadIdx.x >> 2, seg = threadIdx.x & 3;
        const uint4* ks4 = (const uint4*)(Kv + (size_t)(kb * 64 + row) * D_KV +
                                          kvh * HEAD_DIM + seg * 32);
        uint4* kd = (uint4*)&Kt[row * 136 + seg * 32];
        kd[0] = ks4[0]; kd[1] = ks4[1]; kd[2] = ks4[2]; kd[3] = ks4[3];
      }
#endif
      // ---- stage V tile transposed (vector global loads, scalar LDS scatter) ----
      int row = threadIdx.x >> 2, seg = threadIdx.x & 3;
      union { uint4 q[4]; unsigned short u[32]; } vv;
      const uint4* vs4 = (const uint4*)(Vv + (size_t)(kb * 64 + row) * D_KV +
                                        kvh * HEAD_DIM + seg * 32);
      vv.q[0] = vs4[0]; vv.q[1] = vs4[1]; vv.q[2] = vs4[2]; vv.q[3] = vs4[3];
#pragma unroll
      for (int i = 0; i < 32; ++i) Vt[(seg * 32 + i) * 72 + row] = vv.u[i];
    }
#if HAVE_ASYNC && !HAVE_TDM
    __builtin_amdgcn_s_wait_asynccnt(0);
#endif
    __syncthreads();

    // scores: 16 x 64 tile = 4 n-tiles x 4 K-chunks of WMMA
    v8f st[4];
#pragma unroll
    for (int nt = 0; nt < 4; ++nt) {
      v8f acc = {};
#pragma unroll
      for (int kc = 0; kc < 4; ++kc) {
        Frag bk;
        const unsigned short* kp = &Kt[(nt * 16 + r) * 136 + kc * 32];
        bk.q[0] = *(const uint4*)(kp + hi * 8);
        bk.q[1] = *(const uint4*)(kp + 16 + hi * 8);
        acc = wmma_bf16(qa[kc], bk, acc);
      }
      st[nt] = acc;
    }

    // scale + causal mask (C layout: lane = key col, vgpr j = query row)
    const float scale = 0.08838834764831845f;  // 1/sqrt(128)
#pragma unroll
    for (int nt = 0; nt < 4; ++nt) {
      int key = kb * 64 + nt * 16 + r;
#pragma unroll
      for (int j = 0; j < 8; ++j) {
        int qi = qr + hi * 8 + j;
        float sv = st[nt][j] * scale;
        st[nt][j] = (key <= qi) ? sv : -1e30f;
      }
    }

    // online softmax (reductions across the 16-lane half-group)
#pragma unroll
    for (int j = 0; j < 8; ++j) {
      float mx = fmaxf(fmaxf(st[0][j], st[1][j]), fmaxf(st[2][j], st[3][j]));
      mx = rmax16(mx);
      float mn = fmaxf(m[j], mx);
      float alpha = __expf(m[j] - mn);
      m[j] = mn;
      float rs = 0.f;
#pragma unroll
      for (int nt = 0; nt < 4; ++nt) {
        float p = __expf(st[nt][j] - mn);
        st[nt][j] = p;
        rs += p;
      }
      rs = rsum16(rs);
      l[j] = l[j] * alpha + rs;
#pragma unroll
      for (int ot = 0; ot < 8; ++ot) o[ot][j] *= alpha;
    }

    // P: C-layout -> per-wave LDS scratch -> A-layout bf16 fragments
#pragma unroll
    for (int nt = 0; nt < 4; ++nt)
#pragma unroll
      for (int j = 0; j < 8; ++j)
        pw[(hi * 8 + j) * 72 + nt * 16 + r] = f2bf(st[nt][j]);
    __builtin_amdgcn_wave_barrier();   // per-wave DS stream is in-order

    Frag pa[2];
#pragma unroll
    for (int kc2 = 0; kc2 < 2; ++kc2) {
      const unsigned short* pp = pw + r * 72 + kc2 * 32;
      pa[kc2].q[0] = *(const uint4*)(pp + hi * 8);
      pa[kc2].q[1] = *(const uint4*)(pp + 16 + hi * 8);
    }

    // O += P(16x64) * V(64x128): 8 n-tiles x 2 K-chunks of WMMA
#pragma unroll
    for (int ot = 0; ot < 8; ++ot)
#pragma unroll
      for (int kc2 = 0; kc2 < 2; ++kc2) {
        Frag bv;
        const unsigned short* vp = &Vt[(ot * 16 + r) * 72 + kc2 * 32];
        bv.q[0] = *(const uint4*)(vp + hi * 8);
        bv.q[1] = *(const uint4*)(vp + 16 + hi * 8);
        o[ot] = wmma_bf16(pa[kc2], bv, o[ot]);
      }
  }

  float inv[8];
#pragma unroll
  for (int j = 0; j < 8; ++j) inv[j] = 1.0f / l[j];
#pragma unroll
  for (int ot = 0; ot < 8; ++ot)
#pragma unroll
    for (int j = 0; j < 8; ++j)
      O[(size_t)(qr + hi * 8 + j) * D_MODEL + h * HEAD_DIM + ot * 16 + r] =
          f2bf(o[ot][j] * inv[j]);
}

// ---------------- host launcher ----------------
extern "C" void kernel_launch(void* const* d_in, const int* in_sizes, int n_in,
                              void* d_out, int out_size, void* d_ws, size_t ws_size,
                              hipStream_t stream) {
  const float* X    = (const float*)d_in[0];
  const float* cosT = (const float*)d_in[1];
  const float* sinT = (const float*)d_in[2];
  const float* Wq   = (const float*)d_in[3];
  const float* Wk   = (const float*)d_in[4];
  const float* Wv   = (const float*)d_in[5];
  const float* Wo   = (const float*)d_in[6];
  float* out = (float*)d_out;

  char* ws = (char*)d_ws;
  size_t off = 0;
  auto take = [&](size_t bytes) -> char* {
    char* p = ws + off;
    off += (bytes + 255) & ~(size_t)255;
    return p;
  };
  unsigned short* Xbf  = (unsigned short*)take((size_t)S_LEN * D_MODEL * 2);
  unsigned short* Wqbf = (unsigned short*)take((size_t)D_MODEL * D_MODEL * 2);
  unsigned short* Wkbf = (unsigned short*)take((size_t)D_KV * D_MODEL * 2);
  unsigned short* Wvbf = (unsigned short*)take((size_t)D_KV * D_MODEL * 2);
  unsigned short* Wobf = (unsigned short*)take((size_t)D_MODEL * D_MODEL * 2);
  float* Qf = (float*)take((size_t)S_LEN * D_MODEL * 4);
  float* Kf = (float*)take((size_t)S_LEN * D_KV * 4);
  unsigned short* Qbf = (unsigned short*)take((size_t)S_LEN * D_MODEL * 2);
  unsigned short* Kbf = (unsigned short*)take((size_t)S_LEN * D_KV * 2);
  unsigned short* Vbf = (unsigned short*)take((size_t)S_LEN * D_KV * 2);
  unsigned short* Abf = (unsigned short*)take((size_t)S_LEN * D_MODEL * 2);

  auto cvt = [&](const float* src, unsigned short* dst, int n) {
    cvt_kernel<<<(n + 255) / 256, 256, 0, stream>>>(src, dst, n);
  };
  cvt(X,  Xbf,  S_LEN * D_MODEL);
  cvt(Wq, Wqbf, D_MODEL * D_MODEL);
  cvt(Wk, Wkbf, D_KV * D_MODEL);
  cvt(Wv, Wvbf, D_KV * D_MODEL);
  cvt(Wo, Wobf, D_MODEL * D_MODEL);

  // projections: Q (f32 out, needs RoPE), K (f32 out), V (bf16 out, direct)
  gemm_bf16_kernel<<<dim3(D_MODEL / 128, S_LEN / 128), 256, 0, stream>>>(
      Xbf, Wqbf, Qf, D_MODEL, D_MODEL, 0);
  gemm_bf16_kernel<<<dim3(D_KV / 128, S_LEN / 128), 256, 0, stream>>>(
      Xbf, Wkbf, Kf, D_MODEL, D_KV, 0);
  gemm_bf16_kernel<<<dim3(D_KV / 128, S_LEN / 128), 256, 0, stream>>>(
      Xbf, Wvbf, Vbf, D_MODEL, D_KV, 1);

  rope_kernel<<<(S_LEN * N_HEADS * 64 + 255) / 256, 256, 0, stream>>>(
      Qf, cosT, sinT, Qbf, N_HEADS);
  rope_kernel<<<(S_LEN * N_KV * 64 + 255) / 256, 256, 0, stream>>>(
      Kf, cosT, sinT, Kbf, N_KV);

  attn_kernel<<<dim3(S_LEN / 128, N_HEADS), 256, 0, stream>>>(Qbf, Kbf, Vbf, Abf);

  gemm_bf16_kernel<<<dim3(D_MODEL / 128, S_LEN / 128), 256, 0, stream>>>(
      Abf, Wobf, out, D_MODEL, D_MODEL, 0);
}
